// MultitaskModel_51264729645715
// MI455X (gfx1250) — matrix operand order repaired and checked by reading, not compiled, not run on hardware
//
#include <hip/hip_runtime.h>
#include <hip/hip_bf16.h>

// Problem dims (fixed by the reference)
#define B_  32
#define S_  48
#define T_  48
#define E_  512
#define H_  1024
#define G3  3072      // 3*H
#define V_  32000

typedef __bf16 bf16;
typedef __attribute__((ext_vector_type(16))) __bf16 v16bf;
typedef __attribute__((ext_vector_type(8)))  float  v8f;
typedef __attribute__((ext_vector_type(4)))  unsigned int v4u;
typedef __attribute__((ext_vector_type(8)))  int v8i;
typedef __attribute__((ext_vector_type(4)))  int v4i;

union FragBF { v16bf v; uint4 q[2]; };

#define WMMA_BF16(a, b, c) \
  __builtin_amdgcn_wmma_f32_16x16x32_bf16(false, (a), false, (b), (short)0, (c), false, false)

// ---------------------------------------------------------------------------
// LDS A-tile staging (32 rows x K bf16) via the Tensor Data Mover.
// TDM pad: 4 dwords inserted after every 256 dwords (pad_interval=7,
// pad_amount=3) -> breaks the 2048B row-stride bank aliasing while keeping
// 16B alignment for ds_load_b128.  Padded size for K=1024:
// 16384 dw + 64 chunks * 4 dw = 16640 dw = 33280 bf16.
// A 256-dword pad chunk == a 512-element span of one row (K%512==0), so the
// layout is exactly linear inside each 512-element K-chunk.
// ---------------------------------------------------------------------------
#define LDS_A_ELEMS 33280

__device__ inline void stage_a_tile(const bf16* __restrict__ gA, bf16* lds, int K) {
#if __has_builtin(__builtin_amdgcn_tensor_load_to_lds)
  if ((threadIdx.x >> 5) == 0) {   // wave 0 issues the DMA (EXEC ignored by TDM)
    const unsigned long long ga = (unsigned long long)(uintptr_t)gA;
    const unsigned lds_off = (unsigned)(uintptr_t)lds;   // low 32 bits = LDS byte offset
    v4u g0;
    g0.x = 1u;                                            // count=1, user mode, no gather
    g0.y = lds_off;                                       // lds_addr
    g0.z = (unsigned)ga;                                  // global_addr[31:0]
    g0.w = (unsigned)((ga >> 32) & 0x01FFFFFFull) | 0x80000000u; // addr[56:32] | type=2
    v8i g1;
    g1[0] = 0x07D10000;                  // data_size=2B, pad_en, interval=256dw, amount=4dw
    g1[1] = (K & 0xFFFF) << 16;          // tensor_dim0 low16 (atomic_barrier_addr=0)
    g1[2] = ((K >> 16) & 0xFFFF) | (32 << 16); // tensor_dim0 hi | tensor_dim1=32 low
    g1[3] = (K << 16);                   // tensor_dim1 hi (0) | tile_dim0=K
    g1[4] = 32;                          // tile_dim1=32, tile_dim2=0
    g1[5] = K;                           // tensor_dim0_stride low32
    g1[6] = 0;
    g1[7] = 0;
    v4i z4 = {0, 0, 0, 0};
#if defined(__clang_major__) && (__clang_major__ >= 23)
    v8i z8 = {0, 0, 0, 0, 0, 0, 0, 0};
    __builtin_amdgcn_tensor_load_to_lds(g0, g1, z4, z4, z8, 0);
#else
    __builtin_amdgcn_tensor_load_to_lds(g0, g1, z4, z4, 0);
#endif
    __builtin_amdgcn_s_wait_tensorcnt(0);
  }
#else
  // Fallback: cooperative padded copy
  const unsigned* src = (const unsigned*)gA;
  unsigned* dst = (unsigned*)lds;
  const int total = 32 * (K >> 1);
  for (int dw = threadIdx.x; dw < total; dw += blockDim.x)
    dst[dw + ((dw >> 8) << 2)] = src[dw];
#endif
  __syncthreads();
}

// Padded LDS address of element (row m, col k) of the staged 32xK tile.
// Evaluated once per 512-element K-chunk; linear within the chunk.
__device__ inline const bf16* lds_row_ptr(const bf16* lds, int K, int m, int k) {
  int dw = m * (K >> 1) + (k >> 1);
  dw += (dw >> 8) << 2;                 // mirror TDM padding
  return (const bf16*)((const char*)lds + ((size_t)dw << 2));
}

// A fragment (16x32 bf16) at linear offset kk from a per-chunk base pointer.
// Layout per ISA 7.12.2: lanes 0-15 -> M, k(i) = (i>=8?16:0) + half*8 + (i&7);
// the base pointer already folds in half*8, so the two 16B halves sit at
// +kk and +kk+16 elements (never crossing a pad boundary for kk%32==0).
__device__ inline v16bf load_a_frag_chunk(const bf16* pbase, int kk) {
  FragBF f;
  f.q[0] = *(const uint4*)(pbase + kk);
  f.q[1] = *(const uint4*)(pbase + kk + 16);
  return f.v;
}

// B fragment (32x16): W row-major [N, K] -> column n of B is W row n,
// contiguous in K.  lanes 0-15 hold K=0..7,16..23 (n=lane), lanes 16-31 K=8..15,24..31.
__device__ inline v16bf load_b_frag(const bf16* __restrict__ Wrows, int ldb, int kb) {
  const int lane = threadIdx.x & 31;
  const int n    = lane & 15;
  const int half = lane >> 4;
  const bf16* p = Wrows + (size_t)n * ldb + kb + half * 8;
  FragBF f;
  f.q[0] = *(const uint4*)(p);
  f.q[1] = *(const uint4*)(p + 16);
  return f.v;
}

__device__ inline float fast_sigmoid(float x) { return 1.0f / (1.0f + __expf(-x)); }
__device__ inline float fast_tanh(float x) {
  float e = __expf(2.0f * x);
  return (e - 1.0f) / (e + 1.0f);
}

// ---------------------------------------------------------------------------
// f32 -> bf16 conversion (weights)
// ---------------------------------------------------------------------------
__global__ void cvt_bf16_kernel(const float* __restrict__ in, bf16* __restrict__ out, int n) {
  int i = (blockIdx.x * blockDim.x + threadIdx.x) * 4;
  if (i + 3 < n) {
    float4 v = *(const float4*)(in + i);
    out[i + 0] = (bf16)v.x;
    out[i + 1] = (bf16)v.y;
    out[i + 2] = (bf16)v.z;
    out[i + 3] = (bf16)v.w;
  } else {
    for (; i < n; ++i) out[i] = (bf16)in[i];
  }
}

// ---------------------------------------------------------------------------
// Embedding gather + bf16 convert
// ---------------------------------------------------------------------------
__global__ void gather_emb_kernel(const int* __restrict__ tok,
                                  const float* __restrict__ emb,
                                  bf16* __restrict__ out) {
  const int row = blockIdx.x;
  const int tk  = tok[row];
  const float* src = emb + (size_t)tk * E_;
  bf16* dst        = out + (size_t)row * E_;
  for (int i = threadIdx.x * 4; i < E_; i += blockDim.x * 4) {
    float4 v = *(const float4*)(src + i);
    dst[i + 0] = (bf16)v.x;
    dst[i + 1] = (bf16)v.y;
    dst[i + 2] = (bf16)v.z;
    dst[i + 3] = (bf16)v.w;
  }
}

__global__ void init_h_kernel(float* __restrict__ hf, bf16* __restrict__ hb, int n) {
  int i = blockIdx.x * blockDim.x + threadIdx.x;
  if (i < n) { hf[i] = 0.0f; hb[i] = (bf16)0.0f; }
}

// ---------------------------------------------------------------------------
// bf16 WMMA GEMM:  C[M,N] = A[M,K] * W[N,K]^T + bias[N]
// A tile (32 rows x K) staged to LDS once per block by the TDM.
// Block: 128 threads = 4 waves; each wave owns a 32x64 tile
// (2 M-tiles x 4 N-subtiles = 8 WMMA per K-step, every B frag reused 2x,
// A frags come from LDS with chunk-linear addressing).
// Requires M%32==0, N%256==0, K%512==0, K<=1024.  grid = (N/256, M/32)
// ntC != 0 -> non-temporal C stores (streaming output, keep L2 for weights).
// ---------------------------------------------------------------------------
__global__ __launch_bounds__(128) void gemm_bf16_kernel(
    const bf16* __restrict__ A, const bf16* __restrict__ W,
    const float* __restrict__ bias, float* __restrict__ C,
    int M, int N, int K, int ntC) {
  __shared__ __align__(16) bf16 lsA[LDS_A_ELEMS];
  const int wave  = threadIdx.x >> 5;
  const int lane  = threadIdx.x & 31;
  const int half  = lane >> 4;
  const int mblk  = blockIdx.y * 32;
  const int nbase = blockIdx.x * 256 + wave * 64;

  stage_a_tile(A + (size_t)mblk * K, lsA, K);

  v8f a00 = {}, a01 = {}, a02 = {}, a03 = {};   // rows mblk..mblk+15
  v8f a10 = {}, a11 = {}, a12 = {}, a13 = {};   // rows mblk+16..mblk+31
  const bf16* W0 = W + (size_t)(nbase +  0) * K;
  const bf16* W1 = W + (size_t)(nbase + 16) * K;
  const bf16* W2 = W + (size_t)(nbase + 32) * K;
  const bf16* W3 = W + (size_t)(nbase + 48) * K;

  for (int kc = 0; kc < K; kc += 512) {
    const bf16* pa0 = lds_row_ptr(lsA, K, (lane & 15),      kc + half * 8);
    const bf16* pa1 = lds_row_ptr(lsA, K, 16 + (lane & 15), kc + half * 8);
    for (int kk = 0; kk < 512; kk += 32) {
      const int kb = kc + kk;
      v16bf fa0 = load_a_frag_chunk(pa0, kk);
      v16bf fa1 = load_a_frag_chunk(pa1, kk);
      v16bf b0 = load_b_frag(W0, K, kb);
      a00 = WMMA_BF16(fa0, b0, a00);
      a10 = WMMA_BF16(fa1, b0, a10);
      v16bf b1 = load_b_frag(W1, K, kb);
      a01 = WMMA_BF16(fa0, b1, a01);
      a11 = WMMA_BF16(fa1, b1, a11);
      v16bf b2 = load_b_frag(W2, K, kb);
      a02 = WMMA_BF16(fa0, b2, a02);
      a12 = WMMA_BF16(fa1, b2, a12);
      v16bf b3 = load_b_frag(W3, K, kb);
      a03 = WMMA_BF16(fa0, b3, a03);
      a13 = WMMA_BF16(fa1, b3, a13);
    }
  }

  const int n  = lane & 15;
  const int mo = half * 8;
  const float bs0 = bias[nbase +  0 + n];
  const float bs1 = bias[nbase + 16 + n];
  const float bs2 = bias[nbase + 32 + n];
  const float bs3 = bias[nbase + 48 + n];
  if (ntC) {
#pragma unroll
    for (int r = 0; r < 8; ++r) {
      const size_t row0 = (size_t)(mblk + mo + r) * N + nbase;
      const size_t row1 = (size_t)(mblk + 16 + mo + r) * N + nbase;
      __builtin_nontemporal_store(a00[r] + bs0, &C[row0 +  0 + n]);
      __builtin_nontemporal_store(a01[r] + bs1, &C[row0 + 16 + n]);
      __builtin_nontemporal_store(a02[r] + bs2, &C[row0 + 32 + n]);
      __builtin_nontemporal_store(a03[r] + bs3, &C[row0 + 48 + n]);
      __builtin_nontemporal_store(a10[r] + bs0, &C[row1 +  0 + n]);
      __builtin_nontemporal_store(a11[r] + bs1, &C[row1 + 16 + n]);
      __builtin_nontemporal_store(a12[r] + bs2, &C[row1 + 32 + n]);
      __builtin_nontemporal_store(a13[r] + bs3, &C[row1 + 48 + n]);
    }
  } else {
#pragma unroll
    for (int r = 0; r < 8; ++r) {
      const size_t row0 = (size_t)(mblk + mo + r) * N + nbase;
      const size_t row1 = (size_t)(mblk + 16 + mo + r) * N + nbase;
      C[row0 +  0 + n] = a00[r] + bs0;
      C[row0 + 16 + n] = a01[r] + bs1;
      C[row0 + 32 + n] = a02[r] + bs2;
      C[row0 + 48 + n] = a03[r] + bs3;
      C[row1 +  0 + n] = a10[r] + bs0;
      C[row1 + 16 + n] = a11[r] + bs1;
      C[row1 + 32 + n] = a12[r] + bs2;
      C[row1 + 48 + n] = a13[r] + bs3;
    }
  }
}

// ---------------------------------------------------------------------------
// Fused GRU step (one launch per timestep).  h_prev (32xH bf16) staged to LDS
// by the TDM; r/z/n chunks of 64 columns per block computed with WMMA, gate
// math applied in-register.  dec==0: torch GRU; dec==1: conditional GRU.
// grid = H/64 = 16 blocks, 256 threads (8 waves: 2 M-tiles x 4 N-subtiles).
// ---------------------------------------------------------------------------
__global__ __launch_bounds__(256) void gru_step_kernel(
    const bf16*  __restrict__ h_bf,    // [B, H] bf16 (prev)
    const float* __restrict__ h_f,     // [B, H] f32  (prev)
    const bf16*  __restrict__ Wh,      // [3H, H] bf16
    const float* __restrict__ bh,      // [3H]
    const float* __restrict__ xg,      // [B, seqlen, 3H] (includes input bias)
    const float* __restrict__ pg,      // [B, 3H] peephole (dec) or unused
    float* __restrict__ h_f_out,
    bf16*  __restrict__ h_bf_out,
    bf16*  __restrict__ trg_out,       // dec: [B, T, H] bf16 history; enc: null
    int t, int seqlen, int dec) {
  __shared__ __align__(16) bf16 lsH[LDS_A_ELEMS];
  const int wave  = threadIdx.x >> 5;
  const int lane  = threadIdx.x & 31;
  const int half  = lane >> 4;
  const int mbase = (wave & 1) * 16;                    // batch rows
  const int jbase = blockIdx.x * 64 + (wave >> 1) * 16; // hidden columns

  stage_a_tile(h_bf, lsH, H_);

  v8f ar = {}, az = {}, an = {};
  const bf16* Wr = Wh + (size_t)(0 * H_ + jbase) * H_;
  const bf16* Wz = Wh + (size_t)(1 * H_ + jbase) * H_;
  const bf16* Wn = Wh + (size_t)(2 * H_ + jbase) * H_;

  for (int kc = 0; kc < H_; kc += 512) {
    const bf16* pa = lds_row_ptr(lsH, H_, mbase + (lane & 15), kc + half * 8);
    for (int kk = 0; kk < 512; kk += 32) {
      const int kb = kc + kk;
      v16bf a  = load_a_frag_chunk(pa, kk);
      v16bf br = load_b_frag(Wr, H_, kb);
      ar = WMMA_BF16(a, br, ar);
      v16bf bz = load_b_frag(Wz, H_, kb);
      az = WMMA_BF16(a, bz, az);
      v16bf bn = load_b_frag(Wn, H_, kb);
      an = WMMA_BF16(a, bn, an);
    }
  }

  const int n  = lane & 15;
  const int mo = half * 8;
  const int j  = jbase + n;
  const float bhr = bh[j], bhz = bh[H_ + j], bhn = bh[2 * H_ + j];

#pragma unroll
  for (int r = 0; r < 8; ++r) {
    const int b = mbase + mo + r;
    const float* xrow = xg + ((size_t)b * seqlen + t) * G3;
    const float hr = ar[r] + bhr;
    const float hz = az[r] + bhz;
    const float hn = an[r] + bhn;
    const float hprev = h_f[(size_t)b * H_ + j];
    float hnew;
    if (!dec) {
      const float rg = fast_sigmoid(xrow[j] + hr);
      const float zg = fast_sigmoid(xrow[H_ + j] + hz);
      const float ng = fast_tanh(xrow[2 * H_ + j] + rg * hn);
      hnew = (1.0f - zg) * ng + zg * hprev;
    } else {
      const float* prow = pg + (size_t)b * G3;
      const float rg = fast_sigmoid(xrow[j] + hr + prow[j]);
      const float zg = fast_sigmoid(xrow[H_ + j] + hz + prow[H_ + j]);
      const float ng = fast_tanh(xrow[2 * H_ + j] + rg * hn + prow[2 * H_ + j]);
      hnew = ng + zg * (hprev - ng);
    }
    h_f_out[(size_t)b * H_ + j]  = hnew;
    const bf16 hb = (bf16)hnew;
    h_bf_out[(size_t)b * H_ + j] = hb;
    if (trg_out) trg_out[((size_t)b * T_ + t) * H_ + j] = hb;
  }
}

// ---------------------------------------------------------------------------
// Host orchestration
// ---------------------------------------------------------------------------
extern "C" void kernel_launch(void* const* d_in, const int* in_sizes, int n_in,
                              void* d_out, int out_size, void* d_ws, size_t ws_size,
                              hipStream_t stream) {
  const int*   input_src = (const int*)  d_in[0];
  const int*   input_trg = (const int*)  d_in[1];
  const float* src_emb   = (const float*)d_in[2];
  const float* trg_emb   = (const float*)d_in[3];
  const float* enc_Wih   = (const float*)d_in[4];
  const float* enc_Whh   = (const float*)d_in[5];
  const float* enc_bih   = (const float*)d_in[6];
  const float* enc_bhh   = (const float*)d_in[7];
  const float* dec_Wi    = (const float*)d_in[8];
  const float* dec_bi    = (const float*)d_in[9];
  const float* dec_Wh    = (const float*)d_in[10];
  const float* dec_bh    = (const float*)d_in[11];
  const float* dec_Wp    = (const float*)d_in[12];
  const float* dec_bp    = (const float*)d_in[13];
  const float* out_W     = (const float*)d_in[14];
  const float* out_b     = (const float*)d_in[15];
  float* logits = (float*)d_out;

  char* ws = (char*)d_ws;
  size_t off = 0;
  auto carve = [&](size_t bytes) -> void* {
    void* p = ws + off;
    off = (off + bytes + 255) & ~(size_t)255;
    return p;
  };

  bf16* wEncWih = (bf16*)carve((size_t)G3 * E_ * 2);
  bf16* wEncWhh = (bf16*)carve((size_t)G3 * H_ * 2);
  bf16* wDecWi  = (bf16*)carve((size_t)G3 * E_ * 2);
  bf16* wDecWh  = (bf16*)carve((size_t)G3 * H_ * 2);
  bf16* wDecWp  = (bf16*)carve((size_t)G3 * H_ * 2);
  bf16* wOutW   = (bf16*)carve((size_t)V_ * H_ * 2);
  bf16* xemb    = (bf16*)carve((size_t)B_ * S_ * E_ * 2);
  bf16* yemb    = (bf16*)carve((size_t)B_ * T_ * E_ * 2);
  float* xg     = (float*)carve((size_t)B_ * S_ * G3 * 4);
  float* ig     = (float*)carve((size_t)B_ * T_ * G3 * 4);
  float* pg     = (float*)carve((size_t)B_ * G3 * 4);
  float* hF[2]  = { (float*)carve((size_t)B_ * H_ * 4),
                    (float*)carve((size_t)B_ * H_ * 4) };
  bf16*  hB[2]  = { (bf16*)carve((size_t)B_ * H_ * 2),
                    (bf16*)carve((size_t)B_ * H_ * 2) };
  bf16* trgH    = (bf16*)carve((size_t)B_ * T_ * H_ * 2);

  auto cvt = [&](const float* in, bf16* out, int n) {
    cvt_bf16_kernel<<<(n / 4 + 255) / 256, 256, 0, stream>>>(in, out, n);
  };
  cvt(enc_Wih, wEncWih, G3 * E_);
  cvt(enc_Whh, wEncWhh, G3 * H_);
  cvt(dec_Wi,  wDecWi,  G3 * E_);
  cvt(dec_Wh,  wDecWh,  G3 * H_);
  cvt(dec_Wp,  wDecWp,  G3 * H_);
  cvt(out_W,   wOutW,   V_ * H_);

  gather_emb_kernel<<<B_ * S_, 128, 0, stream>>>(input_src, src_emb, xemb);
  gather_emb_kernel<<<B_ * T_, 128, 0, stream>>>(input_trg, trg_emb, yemb);

  init_h_kernel<<<(B_ * H_ + 255) / 256, 256, 0, stream>>>(hF[0], hB[0], B_ * H_);

  // xg = xemb * enc_Wih^T + enc_bih   [1536, 3072]  (re-read by GRU -> RT stores)
  gemm_bf16_kernel<<<dim3(G3 / 256, (B_ * S_) / 32), 128, 0, stream>>>(
      xemb, wEncWih, enc_bih, xg, B_ * S_, G3, E_, 0);
  // ig = yemb * dec_Wi^T + dec_bi     [1536, 3072]
  gemm_bf16_kernel<<<dim3(G3 / 256, (B_ * T_) / 32), 128, 0, stream>>>(
      yemb, wDecWi, dec_bi, ig, B_ * T_, G3, E_, 0);

  // Encoder recurrence
  int cur = 0;
  for (int t = 0; t < S_; ++t) {
    int nxt = cur ^ 1;
    gru_step_kernel<<<H_ / 64, 256, 0, stream>>>(
        hB[cur], hF[cur], wEncWhh, enc_bhh, xg, nullptr,
        hF[nxt], hB[nxt], nullptr, t, S_, 0);
    cur = nxt;
  }

  // Peephole: pg = ctx * dec_Wp^T + dec_bp   [32, 3072]
  gemm_bf16_kernel<<<dim3(G3 / 256, B_ / 32), 128, 0, stream>>>(
      hB[cur], wDecWp, dec_bp, pg, B_, G3, H_, 0);

  // Decoder recurrence (h0 = ctx = encoder final state)
  for (int t = 0; t < T_; ++t) {
    int nxt = cur ^ 1;
    gru_step_kernel<<<H_ / 64, 256, 0, stream>>>(
        hB[cur], hF[cur], wDecWh, dec_bh, ig, pg,
        hF[nxt], hB[nxt], trgH, t, T_, 1);
    cur = nxt;
  }

  // logits = trg_h * out_W^T + out_b   [1536, 32000]  (write-once -> NT stores)
  gemm_bf16_kernel<<<dim3(V_ / 256, (B_ * T_) / 32), 128, 0, stream>>>(
      trgH, wOutW, out_b, logits, B_ * T_, V_, H_, 1);
}
